// GPT_OSS_MLP_74105365725337
// MI455X (gfx1250) — matrix operand order repaired and checked by reading, not compiled, not run on hardware
//
#include <hip/hip_runtime.h>
#include <hip/hip_bf16.h>

typedef __attribute__((ext_vector_type(16))) _Float16 v16h;
typedef __attribute__((ext_vector_type(8)))  _Float16 v8h;
typedef __attribute__((ext_vector_type(8)))  float    v8f;
typedef __attribute__((ext_vector_type(4)))  float    v4f;

#define OSS_ALPHA 1.702f
#define OSS_LIMIT 7.0f

constexpr int Hdim = 2880;   // hidden size (K of gate/up, N of down)
constexpr int Idim = 2880;   // intermediate size (N of gate/up, K of down)
constexpr int Tdim = 512;    // tokens (M)

constexpr int BM = 128;      // block M tile
constexpr int BN = 64;       // block N tile
constexpr int BK = 32;       // K step (f16 WMMA K)
constexpr int LDSS = 40;     // LDS row stride in halves (32 K + 8 pad, 80B rows)

// ---- WMMA fragment helpers -------------------------------------------------

// A-matrix f16 16x32 layout (ISA 7.12.2):
//   lanes 0-15 : row M=lane,    halves = K[0..7]  then K[16..23]
//   lanes 16-31: row M=lane-16, halves = K[8..15] then K[24..31]
__device__ inline v16h load_a_f32(const float* __restrict__ rowk0, int laneHalf) {
  const float* p = rowk0 + laneHalf * 8;          // +8 K for upper half-wave
  v4f f0 = *(const v4f*)(p);
  v4f f1 = *(const v4f*)(p + 4);
  v4f f2 = *(const v4f*)(p + 16);
  v4f f3 = *(const v4f*)(p + 20);
  v16h a;
#pragma unroll
  for (int i = 0; i < 4; ++i) {
    a[i]      = (_Float16)f0[i];
    a[4 + i]  = (_Float16)f1[i];
    a[8 + i]  = (_Float16)f2[i];
    a[12 + i] = (_Float16)f3[i];
  }
  return a;
}

__device__ inline v16h load_a_f16(const _Float16* __restrict__ rowk0, int laneHalf) {
  const _Float16* p = rowk0 + laneHalf * 8;
  v8h lo = *(const v8h*)(p);        // K[off..off+7]
  v8h hi = *(const v8h*)(p + 16);   // K[off+16..off+23]
  v16h a;
#pragma unroll
  for (int i = 0; i < 8; ++i) { a[i] = lo[i]; a[8 + i] = hi[i]; }
  return a;
}

// B-matrix f16 32x16 layout (ISA 7.12.4 pattern):
//   lanes 0-15 : col N=lane,    halves = K[0..7]  then K[16..23]
//   lanes 16-31: col N=lane-16, halves = K[8..15] then K[24..31]
// LDS tile is stored transposed: lds[n * LDSS + k], so this is contiguous.
__device__ inline v16h load_b_lds(const _Float16* lds, int col, int laneHalf) {
  const _Float16* p = lds + col * LDSS + laneHalf * 8;
  v8h lo = *(const v8h*)(p);
  v8h hi = *(const v8h*)(p + 16);
  v16h b;
#pragma unroll
  for (int i = 0; i < 8; ++i) { b[i] = lo[i]; b[8 + i] = hi[i]; }
  return b;
}

// Stage a 32(K) x 64(N) f32 weight tile into LDS as f16, transposed to [N][K].
// Each thread handles a K-pair for 4 consecutive columns:
//   2 coalesced global b128 loads (non-temporal: weights are streamed once),
//   4 packed b32 LDS stores (halves k,k+1 contiguous in the [N][K] layout).
// Also prefetches the tile PF steps ahead.
__device__ inline void stage_b_tile(const float* __restrict__ W, int N,
                                    int k0, int nb, _Float16* lds, int kmax) {
  const int tid = threadIdx.x;          // 0..255
  const int kp  = tid >> 4;             // 0..15 -> k = 2*kp
  const int n   = (tid & 15) * 4;       // 0..60
  const float* p0 = W + (size_t)(k0 + 2 * kp + 0) * N + nb + n;
  const float* p1 = W + (size_t)(k0 + 2 * kp + 1) * N + nb + n;
  v4f w0 = __builtin_nontemporal_load((const v4f*)p0);
  v4f w1 = __builtin_nontemporal_load((const v4f*)p1);
  const int pfk = k0 + 2 * BK;
  if (pfk < kmax)
    __builtin_prefetch(W + (size_t)(pfk + 2 * kp) * N + nb + n, 0, 1);
  union { _Float16 h[2]; unsigned u; } pk;
#pragma unroll
  for (int j = 0; j < 4; ++j) {
    pk.h[0] = (_Float16)w0[j];
    pk.h[1] = (_Float16)w1[j];
    *(unsigned*)&lds[(n + j) * LDSS + 2 * kp] = pk.u;
  }
}

__device__ inline v8f wmma16x16x32(v16h a, v16h b, v8f c) {
  return __builtin_amdgcn_wmma_f32_16x16x32_f16(
      /*neg_a=*/false, a, /*neg_b=*/false, b,
      /*c_mod=*/(short)0, c, /*reuse_a=*/false, /*reuse_b=*/false);
}

// ---- Stage 0: per-token routing weight -------------------------------------

__global__ void tokw_kernel(const int* __restrict__ mask,
                            const float* __restrict__ rw,
                            float* __restrict__ tokw) {
  int t = blockIdx.x * blockDim.x + threadIdx.x;
  if (t < Tdim) {
    float s = 0.f;
#pragma unroll
    for (int j = 0; j < 4; ++j)
      s += (float)mask[j * Tdim + t] * rw[t * 4 + j];
    tokw[t] = s;
  }
}

// ---- Stage 1: fused gate/up GEMMs + clamped GLU -> h (f16) -----------------

__global__ __launch_bounds__(256) void gateup_kernel(
    const float* __restrict__ X,  const float* __restrict__ Wg,
    const float* __restrict__ Bg, const float* __restrict__ Wu,
    const float* __restrict__ Bu, _Float16* __restrict__ Hout) {
  __shared__ _Float16 ldsG[2][BN * LDSS];
  __shared__ _Float16 ldsU[2][BN * LDSS];

  const int lane     = threadIdx.x & 31;
  const int wave     = threadIdx.x >> 5;
  const int wm       = (wave & 3) << 5;       // wave M offset: 0,32,64,96
  const int wn       = (wave >> 2) << 5;      // wave N offset: 0,32
  const int laneHalf = (lane >> 4) & 1;
  const int lane15   = lane & 15;
  const int mBase    = blockIdx.y * BM;
  const int nBase    = blockIdx.x * BN;

  v8f accg[2][2] = {{{0.f}}};
  v8f accu[2][2] = {{{0.f}}};

  // Prologue: stage first K tile
  stage_b_tile(Wg, Idim, 0, nBase, ldsG[0], Hdim);
  stage_b_tile(Wu, Idim, 0, nBase, ldsU[0], Hdim);
  __syncthreads();

  int cur = 0;
  for (int k0 = 0; k0 < Hdim; k0 += BK) {
    const int nxt = cur ^ 1;
    // Stage next K tile into the other buffer (overlaps with WMMAs below)
    if (k0 + BK < Hdim) {
      stage_b_tile(Wg, Idim, k0 + BK, nBase, ldsG[nxt], Hdim);
      stage_b_tile(Wu, Idim, k0 + BK, nBase, ldsU[nxt], Hdim);
    }

    v16h a[2], bg[2], bu[2];
#pragma unroll
    for (int mi = 0; mi < 2; ++mi) {
      int row = mBase + wm + mi * 16 + lane15;
      a[mi] = load_a_f32(X + (size_t)row * Hdim + k0, laneHalf);
    }
#pragma unroll
    for (int ni = 0; ni < 2; ++ni) {
      int col = wn + ni * 16 + lane15;
      bg[ni] = load_b_lds(ldsG[cur], col, laneHalf);
      bu[ni] = load_b_lds(ldsU[cur], col, laneHalf);
    }
#pragma unroll
    for (int mi = 0; mi < 2; ++mi)
#pragma unroll
      for (int ni = 0; ni < 2; ++ni) {
        accg[mi][ni] = wmma16x16x32(a[mi], bg[ni], accg[mi][ni]);
        accu[mi][ni] = wmma16x16x32(a[mi], bu[ni], accu[mi][ni]);
      }
    __syncthreads();
    cur = nxt;
  }

  // Epilogue: bias + clamped GLU, write h in f16 (A-operand-ready for stage 2)
#pragma unroll
  for (int ni = 0; ni < 2; ++ni) {
    int col = nBase + wn + ni * 16 + lane15;
    float gb = Bg[col];
    float ub = Bu[col];
#pragma unroll
    for (int mi = 0; mi < 2; ++mi) {
#pragma unroll
      for (int r = 0; r < 8; ++r) {
        int row = mBase + wm + mi * 16 + laneHalf * 8 + r;
        float g = accg[mi][ni][r] + gb;
        float u = accu[mi][ni][r] + ub;
        g = fminf(g, OSS_LIMIT);
        u = fminf(fmaxf(u, -OSS_LIMIT), OSS_LIMIT);
        float glu = g / (1.0f + __expf(-OSS_ALPHA * g));
        float h = glu * (u + 1.0f);
        Hout[(size_t)row * Idim + col] = (_Float16)h;
      }
    }
  }
}

// ---- Stage 2: down GEMM + bias + routing scale + add final -----------------

__global__ __launch_bounds__(256) void down_kernel(
    const _Float16* __restrict__ Hin, const float* __restrict__ Wd,
    const float* __restrict__ Bd,     const float* __restrict__ Fin,
    const float* __restrict__ tokw,   float* __restrict__ Out) {
  __shared__ _Float16 ldsD[2][BN * LDSS];

  const int lane     = threadIdx.x & 31;
  const int wave     = threadIdx.x >> 5;
  const int wm       = (wave & 3) << 5;
  const int wn       = (wave >> 2) << 5;
  const int laneHalf = (lane >> 4) & 1;
  const int lane15   = lane & 15;
  const int mBase    = blockIdx.y * BM;
  const int nBase    = blockIdx.x * BN;

  v8f acc[2][2] = {{{0.f}}};

  stage_b_tile(Wd, Hdim, 0, nBase, ldsD[0], Idim);
  __syncthreads();

  int cur = 0;
  for (int k0 = 0; k0 < Idim; k0 += BK) {
    const int nxt = cur ^ 1;
    if (k0 + BK < Idim)
      stage_b_tile(Wd, Hdim, k0 + BK, nBase, ldsD[nxt], Idim);

    v16h a[2], bd[2];
#pragma unroll
    for (int mi = 0; mi < 2; ++mi) {
      int row = mBase + wm + mi * 16 + lane15;
      a[mi] = load_a_f16(Hin + (size_t)row * Idim + k0, laneHalf);
    }
#pragma unroll
    for (int ni = 0; ni < 2; ++ni)
      bd[ni] = load_b_lds(ldsD[cur], wn + ni * 16 + lane15, laneHalf);
#pragma unroll
    for (int mi = 0; mi < 2; ++mi)
#pragma unroll
      for (int ni = 0; ni < 2; ++ni)
        acc[mi][ni] = wmma16x16x32(a[mi], bd[ni], acc[mi][ni]);
    __syncthreads();
    cur = nxt;
  }

#pragma unroll
  for (int ni = 0; ni < 2; ++ni) {
    int col = nBase + wn + ni * 16 + lane15;
    float db = Bd[col];
#pragma unroll
    for (int mi = 0; mi < 2; ++mi) {
#pragma unroll
      for (int r = 0; r < 8; ++r) {
        int row = mBase + wm + mi * 16 + laneHalf * 8 + r;
        float y = acc[mi][ni][r] + db;
        size_t idx = (size_t)row * Hdim + col;
        Out[idx] = Fin[idx] + y * tokw[row];
      }
    }
  }
}

// ---- Host-side launch ------------------------------------------------------

extern "C" void kernel_launch(void* const* d_in, const int* in_sizes, int n_in,
                              void* d_out, int out_size, void* d_ws, size_t ws_size,
                              hipStream_t stream) {
  const float* X   = (const float*)d_in[0];   // hidden_states [T,H]
  const float* RW  = (const float*)d_in[1];   // routing_weights [T,4]
  const float* FHS = (const float*)d_in[2];   // final_hidden_states [T,H]
  const float* Wg  = (const float*)d_in[3];   // gate_w [H,I]
  const float* Bg  = (const float*)d_in[4];   // gate_b [I]
  const float* Wu  = (const float*)d_in[5];   // up_w [H,I]
  const float* Bu  = (const float*)d_in[6];   // up_b [I]
  const float* Wd  = (const float*)d_in[7];   // down_w [I,H]
  const float* Bd  = (const float*)d_in[8];   // down_b [H]
  const int*   EM  = (const int*)d_in[9];     // expert_mask [4,T]
  float* Out = (float*)d_out;

  // Workspace: h (f16, T*I) then tok_w (f32, T)
  _Float16* Hbuf = (_Float16*)d_ws;
  float* tokw = (float*)((char*)d_ws + (size_t)Tdim * Idim * sizeof(_Float16));

  tokw_kernel<<<(Tdim + 255) / 256, 256, 0, stream>>>(EM, RW, tokw);

  dim3 grid1(Idim / BN, Tdim / BM);   // (45, 4)
  gateup_kernel<<<grid1, 256, 0, stream>>>(X, Wg, Bg, Wu, Bu, Hbuf);

  dim3 grid2(Hdim / BN, Tdim / BM);   // (45, 4)
  down_kernel<<<grid2, 256, 0, stream>>>(Hbuf, Wd, Bd, FHS, tokw, Out);
}
